// GraphSAGE_72541997629469
// MI455X (gfx1250) — matrix operand order repaired and checked by reading, not compiled, not run on hardware
//
#include <hip/hip_runtime.h>
#include <hip/hip_bf16.h>

typedef __attribute__((ext_vector_type(16))) _Float16 v16h;
typedef __attribute__((ext_vector_type(8)))  float    v8f;

#define FDIM 128
#define NEG_SLOPE 0.01f
#define L2EPS 1e-12f

__device__ __forceinline__ float leaky(float v) { return v >= 0.0f ? v : NEG_SLOPE * v; }

// ---------------- WMMA fragment loaders (contiguous k order k = 16h + i, same for A and B) ----------------

// A-matrix 16x32 f16, tile stored row-major [16][128] (LDS). lane = 16*h + m.
// element i: k = kb + h*16 + i
__device__ __forceinline__ v16h load_a_frag(const _Float16* a, int lane, int kb) {
  const int row = lane & 15;
  const int h   = lane >> 4;
  const _Float16* p = a + row * FDIM + kb + h * 16;
  v16h r;
#pragma unroll
  for (int i = 0; i < 8; ++i) { r[i] = p[i]; r[i + 8] = p[i + 8]; }
  return r;
}

// B-matrix 32x16 f16 where B[k][n] = W[n][k], W row-major [128][128] f16.
// lane = 16*h + n; element i: k = kb + h*16 + i  -> 16 contiguous halves (32B load)
__device__ __forceinline__ v16h load_b_frag(const _Float16* w, int nBase, int lane, int kb) {
  const _Float16* p = w + (size_t)(nBase + (lane & 15)) * FDIM + kb + (lane >> 4) * 16;
  return *(const v16h*)p;
}

__device__ __forceinline__ v8f wmma_f16(v16h a, v16h b, v8f c) {
  return __builtin_amdgcn_wmma_f32_16x16x32_f16(false, a, false, b, (short)0, c, false, false);
}

// ---------------- small utility kernels ----------------

__global__ void zero_kernel(float* __restrict__ p, int n) {
  int i = blockIdx.x * blockDim.x + threadIdx.x;
  if (i < n) p[i] = 0.0f;
}

__global__ void cvt_f16_kernel(const float* __restrict__ src, _Float16* __restrict__ dst, int n) {
  int i = blockIdx.x * blockDim.x + threadIdx.x;
  if (i < n) dst[i] = (_Float16)src[i];
}

// split fc weight [128][256] -> A half [128][128] (conv part), B half [128][128] (skip part)
__global__ void split_f16_kernel(const float* __restrict__ src, _Float16* __restrict__ dA,
                                 _Float16* __restrict__ dB) {
  int i = blockIdx.x * blockDim.x + threadIdx.x;
  if (i < FDIM * FDIM) {
    int r = i >> 7, c = i & 127;
    dA[i] = (_Float16)src[r * 256 + c];
    dB[i] = (_Float16)src[r * 256 + 128 + c];
  }
}

// row-normalize: one wave per row, lane handles 4 contiguous floats
__global__ void rownorm_kernel(const float* __restrict__ x, float* __restrict__ xn, int N) {
  int gid = blockIdx.x * blockDim.x + threadIdx.x;
  int row = gid >> 5;
  if (row >= N) return;
  int lane = gid & 31;
  const float4 v = *(const float4*)(x + (size_t)row * FDIM + lane * 4);
  float s = v.x + v.y + v.z + v.w;
#pragma unroll
  for (int m = 16; m >= 1; m >>= 1) s += __shfl_xor(s, m, 32);
  float rinv = (s == 0.0f) ? 0.0f : 1.0f / s;
  float4 o = make_float4(v.x * rinv, v.y * rinv, v.z * rinv, v.w * rinv);
  *(float4*)(xn + (size_t)row * FDIM + lane * 4) = o;
}

// edge scatter-add: one wave per edge, lane handles 4 floats; lane0 bumps degree
__global__ void scatter_kernel(const float* __restrict__ feat, const int* __restrict__ src,
                               const int* __restrict__ dst, float* __restrict__ agg,
                               float* __restrict__ deg, int E) {
  int gid = blockIdx.x * blockDim.x + threadIdx.x;
  int e = gid >> 5;
  if (e >= E) return;
  int lane = gid & 31;
  int s = src[e];
  int d = dst[e];
  const float4 v = *(const float4*)(feat + (size_t)s * FDIM + lane * 4);
  float* o = agg + (size_t)d * FDIM + lane * 4;
  atomicAdd(o + 0, v.x);
  atomicAdd(o + 1, v.y);
  atomicAdd(o + 2, v.z);
  atomicAdd(o + 3, v.w);
  if (lane == 0) atomicAdd(deg + d, 1.0f);
}

// node -> graph pooling (segment sum): one wave per node
__global__ void pool_kernel(const float* __restrict__ h, const int* __restrict__ batch,
                            float* __restrict__ pooled, int N) {
  int gid = blockIdx.x * blockDim.x + threadIdx.x;
  int node = gid >> 5;
  if (node >= N) return;
  int lane = gid & 31;
  int g = batch[node];
  const float4 v = *(const float4*)(h + (size_t)node * FDIM + lane * 4);
  float* o = pooled + (size_t)g * FDIM + lane * 4;
  atomicAdd(o + 0, v.x);
  atomicAdd(o + 1, v.y);
  atomicAdd(o + 2, v.z);
  atomicAdd(o + 3, v.w);
}

// ---------------- fused SAGEConv + l2norm + leaky + concat-FC kernel ----------------
// Workgroup = 256 threads (8 waves), owns a 16-node row block.
// conv: c = mean @ Wl.T + feat @ Wr.T ; h = leaky(l2norm(c))
// fc:   out = leaky([h, feat] @ fcW.T + b) with fcW split into Fa (h part) and Fb (feat part)
__global__ __launch_bounds__(256) void conv_fc_kernel(
    const float* __restrict__ feat, const float* __restrict__ agg, const float* __restrict__ deg,
    const _Float16* __restrict__ Wl, const _Float16* __restrict__ Wr,
    const _Float16* __restrict__ Fa, const _Float16* __restrict__ Fb,
    const float* __restrict__ bias, float* __restrict__ out, int N) {
  __shared__ _Float16 aM[16 * FDIM];  // mean tile (then reused as fc A-tile for conv output)
  __shared__ _Float16 aX[16 * FDIM];  // skip/self tile
  __shared__ float cT[16 * FDIM];     // conv f32 result for l2norm
  __shared__ float rn[16];

  const int rowBase = blockIdx.x * 16;
  const int t = threadIdx.x;
  const int lane = t & 31;
  const int wave = t >> 5;
  const int nBase = wave * 16;

  // load + convert tiles: 16*128 elems each, float4 granularity (512 float4, 2 per thread)
  for (int i = t; i < 16 * (FDIM / 4); i += 256) {
    int r = i >> 5;            // 32 float4 per row
    int c4 = (i & 31) * 4;
    int node = rowBase + r;
    if (node >= N) node = N - 1;
    float invd = 1.0f / fmaxf(deg[node], 1.0f);
    const float4 m4 = *(const float4*)(agg + (size_t)node * FDIM + c4);
    const float4 x4 = *(const float4*)(feat + (size_t)node * FDIM + c4);
    aM[r * FDIM + c4 + 0] = (_Float16)(m4.x * invd);
    aM[r * FDIM + c4 + 1] = (_Float16)(m4.y * invd);
    aM[r * FDIM + c4 + 2] = (_Float16)(m4.z * invd);
    aM[r * FDIM + c4 + 3] = (_Float16)(m4.w * invd);
    aX[r * FDIM + c4 + 0] = (_Float16)x4.x;
    aX[r * FDIM + c4 + 1] = (_Float16)x4.y;
    aX[r * FDIM + c4 + 2] = (_Float16)x4.z;
    aX[r * FDIM + c4 + 3] = (_Float16)x4.w;
  }
  __syncthreads();

  // conv GEMM: 16x16 output tile per wave, K=128 in 4 steps, 2 WMMAs per step
  v8f acc = {};
#pragma unroll
  for (int kb = 0; kb < FDIM; kb += 32) {
    acc = wmma_f16(load_a_frag(aM, lane, kb), load_b_frag(Wl, nBase, lane, kb), acc);
    acc = wmma_f16(load_a_frag(aX, lane, kb), load_b_frag(Wr, nBase, lane, kb), acc);
  }
  const int dn = lane & 15;
  const int dm0 = (lane >> 4) * 8;
#pragma unroll
  for (int r = 0; r < 8; ++r) cT[(dm0 + r) * FDIM + nBase + dn] = acc[r];
  __syncthreads();

  // row-wise L2 norms
  if (t < 16) {
    float s = 0.0f;
    for (int c = 0; c < FDIM; ++c) {
      float v = cT[t * FDIM + c];
      s += v * v;
    }
    rn[t] = fmaxf(sqrtf(s), L2EPS);
  }
  __syncthreads();

  // h = leaky(c / ||c||) -> f16 into aM (fc A-tile); aX keeps the skip features
  for (int i = t; i < 16 * FDIM; i += 256) {
    int r = i >> 7;
    aM[i] = (_Float16)leaky(cT[i] / rn[r]);
  }
  __syncthreads();

  // fc GEMM on concat([h, skip]) via split weights
  v8f acc2 = {};
#pragma unroll
  for (int kb = 0; kb < FDIM; kb += 32) {
    acc2 = wmma_f16(load_a_frag(aM, lane, kb), load_b_frag(Fa, nBase, lane, kb), acc2);
    acc2 = wmma_f16(load_a_frag(aX, lane, kb), load_b_frag(Fb, nBase, lane, kb), acc2);
  }
  float bj = bias[nBase + dn];
#pragma unroll
  for (int r = 0; r < 8; ++r) {
    int m = rowBase + dm0 + r;
    if (m < N) out[(size_t)m * FDIM + nBase + dn] = leaky(acc2[r] + bj);
  }
}

// final: out = l2norm(leaky(pooled @ fc3W.T + b)), one block (128 thr) per graph
__global__ __launch_bounds__(128) void fc3_kernel(const float* __restrict__ pooled,
                                                  const float* __restrict__ W,
                                                  const float* __restrict__ b,
                                                  float* __restrict__ out, int G) {
  __shared__ float row[FDIM];
  __shared__ float sq[FDIM];
  int g = blockIdx.x;
  int j = threadIdx.x;
  row[j] = pooled[(size_t)g * FDIM + j];
  __syncthreads();
  float acc = b[j];
  for (int k = 0; k < FDIM; ++k) acc += row[k] * W[j * FDIM + k];
  acc = leaky(acc);
  sq[j] = acc * acc;
  __syncthreads();
  for (int s = 64; s > 0; s >>= 1) {
    if (j < s) sq[j] += sq[j + s];
    __syncthreads();
  }
  float nrm = fmaxf(sqrtf(sq[0]), L2EPS);
  out[(size_t)g * FDIM + j] = acc / nrm;
}

// ---------------- launcher ----------------

extern "C" void kernel_launch(void* const* d_in, const int* in_sizes, int n_in,
                              void* d_out, int out_size, void* d_ws, size_t ws_size,
                              hipStream_t stream) {
  const float* x     = (const float*)d_in[0];
  const int*   ei    = (const int*)d_in[1];
  const int*   batch = (const int*)d_in[2];
  const float* W1l   = (const float*)d_in[3];
  const float* W1r   = (const float*)d_in[4];
  const float* W2l   = (const float*)d_in[5];
  const float* W2r   = (const float*)d_in[6];
  const float* fc1W  = (const float*)d_in[7];
  const float* fc1b  = (const float*)d_in[8];
  const float* fc2W  = (const float*)d_in[9];
  const float* fc2b  = (const float*)d_in[10];
  const float* fc3W  = (const float*)d_in[11];
  const float* fc3b  = (const float*)d_in[12];

  const int N = in_sizes[0] / FDIM;
  const int E = in_sizes[1] / 2;
  const int G = out_size / FDIM;
  const int* src = ei;
  const int* dst = ei + E;

  // workspace carve-out
  char* ws = (char*)d_ws;
  size_t off = 0;
  auto carve = [&](size_t bytes) {
    void* p = ws + off;
    off = (off + bytes + 255) & ~(size_t)255;
    return p;
  };
  float* xn     = (float*)carve((size_t)N * FDIM * 4);  // x_norm, later reused for h2
  float* agg    = (float*)carve((size_t)N * FDIM * 4);
  float* hbuf   = (float*)carve((size_t)N * FDIM * 4);
  float* deg    = (float*)carve((size_t)N * 4);
  float* pooled = (float*)carve((size_t)G * FDIM * 4);
  _Float16* w1l16 = (_Float16*)carve(FDIM * FDIM * 2);
  _Float16* w1r16 = (_Float16*)carve(FDIM * FDIM * 2);
  _Float16* w2l16 = (_Float16*)carve(FDIM * FDIM * 2);
  _Float16* w2r16 = (_Float16*)carve(FDIM * FDIM * 2);
  _Float16* fc1a  = (_Float16*)carve(FDIM * FDIM * 2);
  _Float16* fc1b16= (_Float16*)carve(FDIM * FDIM * 2);
  _Float16* fc2a  = (_Float16*)carve(FDIM * FDIM * 2);
  _Float16* fc2b16= (_Float16*)carve(FDIM * FDIM * 2);
  (void)ws_size; (void)n_in;

  const int WSQ = FDIM * FDIM;  // 16384
  const int cvtBlocks = (WSQ + 255) / 256;
  // weight conversion to f16
  cvt_f16_kernel<<<cvtBlocks, 256, 0, stream>>>(W1l, w1l16, WSQ);
  cvt_f16_kernel<<<cvtBlocks, 256, 0, stream>>>(W1r, w1r16, WSQ);
  cvt_f16_kernel<<<cvtBlocks, 256, 0, stream>>>(W2l, w2l16, WSQ);
  cvt_f16_kernel<<<cvtBlocks, 256, 0, stream>>>(W2r, w2r16, WSQ);
  split_f16_kernel<<<cvtBlocks, 256, 0, stream>>>(fc1W, fc1a, fc1b16);
  split_f16_kernel<<<cvtBlocks, 256, 0, stream>>>(fc2W, fc2a, fc2b16);

  // preprocess: row-normalize
  rownorm_kernel<<<((size_t)N * 32 + 255) / 256, 256, 0, stream>>>(x, xn, N);

  const int featN = N * FDIM;
  const int edgeBlocks = (int)(((size_t)E * 32 + 255) / 256);
  const int nodeBlocks = (int)(((size_t)N * 32 + 255) / 256);
  const int rowBlocks  = (N + 15) / 16;

  // ---- layer 1 ----
  zero_kernel<<<(featN + 255) / 256, 256, 0, stream>>>(agg, featN);
  zero_kernel<<<(N + 255) / 256, 256, 0, stream>>>(deg, N);
  scatter_kernel<<<edgeBlocks, 256, 0, stream>>>(xn, src, dst, agg, deg, E);
  conv_fc_kernel<<<rowBlocks, 256, 0, stream>>>(xn, agg, deg, w1l16, w1r16, fc1a, fc1b16,
                                                fc1b, hbuf, N);

  // ---- layer 2 ----
  zero_kernel<<<(featN + 255) / 256, 256, 0, stream>>>(agg, featN);
  zero_kernel<<<(N + 255) / 256, 256, 0, stream>>>(deg, N);
  scatter_kernel<<<edgeBlocks, 256, 0, stream>>>(hbuf, src, dst, agg, deg, E);
  conv_fc_kernel<<<rowBlocks, 256, 0, stream>>>(hbuf, agg, deg, w2l16, w2r16, fc2a, fc2b16,
                                                fc2b, xn, N);  // write h2 into xn (reuse)

  // ---- pool + head ----
  zero_kernel<<<(G * FDIM + 255) / 256, 256, 0, stream>>>(pooled, G * FDIM);
  pool_kernel<<<nodeBlocks, 256, 0, stream>>>(xn, batch, pooled, N);
  fc3_kernel<<<G, 128, 0, stream>>>(pooled, fc3W, fc3b, (float*)d_out, G);
}
